// TableWiseLayer_55327768707127
// MI455X (gfx1250) — matrix operand order repaired and checked by reading, not compiled, not run on hardware
//
#include <hip/hip_runtime.h>
#include <hip/hip_bf16.h>
#include <math.h>

// ---------------------------------------------------------------------------
// TableWiseLayer for MI455X (gfx1250):
//   GEMM1 (concat(former,hidden) @ W_down) + bias + erf-GELU   -> seq_out
//   deterministic segment-sum by token_type_ids                -> cell_sum
//   GEMM2 (cell_sum @ W_row) + bias + erf-GELU                 -> cell_out
//   gather rows back per token                                  -> out
// GEMMs: v_wmma_f32_16x16x32_bf16, 128x256 block tile, 8 waves x (64x64),
// register-staged double-buffered LDS (one barrier per K-step).
// ---------------------------------------------------------------------------

typedef __bf16 bf16_t;
typedef __attribute__((ext_vector_type(16))) __bf16 v16bf;
typedef __attribute__((ext_vector_type(8)))  __bf16 v8bf;
typedef __attribute__((ext_vector_type(8)))  float  v8f;
typedef __attribute__((ext_vector_type(4)))  float  v4f;

#define TM 128
#define TN 256
#define TK 32
#define KP 40   // padded LDS row stride in bf16 elems (80B, 16B-aligned)

__device__ __forceinline__ float gelu_erf(float x) {
    return 0.5f * x * (1.0f + erff(x * 0.70710678118654752440f));
}

__device__ __forceinline__ v8bf cvt8_f32_bf16(const float* __restrict__ p) {
    v4f a = *(const v4f*)p;
    v4f b = *(const v4f*)(p + 4);
    v8bf r;
    r[0] = (__bf16)a.x; r[1] = (__bf16)a.y; r[2] = (__bf16)a.z; r[3] = (__bf16)a.w;
    r[4] = (__bf16)b.x; r[5] = (__bf16)b.y; r[6] = (__bf16)b.z; r[7] = (__bf16)b.w;
    return r;
}

// WT[n*K + k] = (bf16) W[k*N + n]   (one-time, coalesced bf16 writes)
__global__ __launch_bounds__(256)
void convert_transpose_kernel(const float* __restrict__ W, bf16_t* __restrict__ WT,
                              int K, int N) {
    int idx = blockIdx.x * 256 + threadIdx.x;   // over N*K, k fastest
    int n = idx / K;
    int k = idx - n * K;
    if (n < N) WT[(size_t)n * K + k] = (__bf16)W[(size_t)k * N + n];
}

// C[M,N] = GELU( A[M,K] @ W[K,N] + bias ), W transposed bf16 as WT[N,K].
// Logical A[row][k] = (k < A0cols) ? A0[row*A0cols+k] : A1[row*A0cols+k-A0cols]
// (A0cols == K for the non-concat case). M mult of 128, N mult of 256,
// K mult of 32, A0cols mult of 32.
__global__ __launch_bounds__(256)
void gemm_bias_gelu_kernel(const float* __restrict__ A0, const float* __restrict__ A1,
                           int A0cols, const bf16_t* __restrict__ WT, int K,
                           const float* __restrict__ bias, float* __restrict__ C,
                           int N) {
    __shared__ __bf16 As[2][TM * KP];   // 2 x 10240 B
    __shared__ __bf16 Bs[2][TN * KP];   // 2 x 20480 B

    const int tid  = threadIdx.x;
    const int lane = tid & 31;
    const int wid  = tid >> 5;              // 8 waves: 2 (M) x 4 (N)
    const int wm   = (wid >> 2) * 64;       // wave M offset in block tile
    const int wn   = (wid & 3) * 64;        // wave N offset in block tile
    const int m0   = blockIdx.x * TM;
    const int n0   = blockIdx.y * TN;

    // A loader: 2 threads per row, 16 fp32 each. B loader: 1 thread/row, 32 bf16.
    const int arow = tid >> 1;
    const int acol = (tid & 1) * 16;
    const int brow = tid;

    const int fm = lane & 15;               // fragment row/col within 16
    const int g  = lane >> 4;               // lane half

    v8f acc[4][4] = {};

    // staging registers (global -> reg -> LDS double buffering)
    v8bf sa0, sa1, sb0, sb1, sb2, sb3;

    const float*  aptr_c = nullptr;
    const bf16_t* bptr_c = nullptr;

#define STAGE(k0)                                                               \
    do {                                                                        \
        int kk = (k0) + acol;               /* TK<=A0cols: never straddles */   \
        aptr_c = (kk < A0cols)                                                  \
            ? (A0 + (size_t)(m0 + arow) * A0cols + kk)                          \
            : (A1 + (size_t)(m0 + arow) * A0cols + (kk - A0cols));              \
        sa0 = cvt8_f32_bf16(aptr_c);                                            \
        sa1 = cvt8_f32_bf16(aptr_c + 8);                                        \
        bptr_c = WT + (size_t)(n0 + brow) * K + (k0);                           \
        sb0 = *(const v8bf*)(bptr_c);                                           \
        sb1 = *(const v8bf*)(bptr_c + 8);                                       \
        sb2 = *(const v8bf*)(bptr_c + 16);                                      \
        sb3 = *(const v8bf*)(bptr_c + 24);                                      \
    } while (0)

#define COMMIT(buf)                                                             \
    do {                                                                        \
        *(v8bf*)(&As[buf][arow * KP + acol    ]) = sa0;                         \
        *(v8bf*)(&As[buf][arow * KP + acol + 8]) = sa1;                         \
        *(v8bf*)(&Bs[buf][brow * KP +  0]) = sb0;                               \
        *(v8bf*)(&Bs[buf][brow * KP +  8]) = sb1;                               \
        *(v8bf*)(&Bs[buf][brow * KP + 16]) = sb2;                               \
        *(v8bf*)(&Bs[buf][brow * KP + 24]) = sb3;                               \
    } while (0)

    STAGE(0);
    COMMIT(0);
    __syncthreads();

    const int nk = K / TK;
    for (int kt = 0; kt < nk; ++kt) {
        const int buf = kt & 1;
        if (kt + 1 < nk) {
            STAGE((kt + 1) * TK);           // global loads overlap WMMAs below
            __builtin_prefetch(bptr_c + TK, 0, 0);   // peek one tile further
        }

        // ---- B fragments (32x16 bf16: lanes 0-15 K=0..15, lanes 16-31 K=16..31)
        v16bf bfrag[4];
#pragma unroll
        for (int j = 0; j < 4; ++j) {
            const __bf16* bp = &Bs[buf][(wn + j * 16 + fm) * KP + g * 16];
            v8bf lo = *(const v8bf*)bp;
            v8bf hi = *(const v8bf*)(bp + 8);
            bfrag[j] = __builtin_shufflevector(lo, hi,
                0, 1, 2, 3, 4, 5, 6, 7, 8, 9, 10, 11, 12, 13, 14, 15);
        }

        // ---- A fragments + 16 WMMAs
        // A 16x32 bf16: lanes 0-15 K=0..15, lanes 16-31 K=16..31 (matches B)
#pragma unroll
        for (int i = 0; i < 4; ++i) {
            const __bf16* ap = &As[buf][(wm + i * 16 + fm) * KP + g * 16];
            v8bf alo = *(const v8bf*)ap;
            v8bf ahi = *(const v8bf*)(ap + 8);
            v16bf a = __builtin_shufflevector(alo, ahi,
                0, 1, 2, 3, 4, 5, 6, 7, 8, 9, 10, 11, 12, 13, 14, 15);
#pragma unroll
            for (int j = 0; j < 4; ++j) {
                acc[i][j] = __builtin_amdgcn_wmma_f32_16x16x32_bf16(
                    false, a, false, bfrag[j], (short)0, acc[i][j], false, false);
            }
        }

        if (kt + 1 < nk) COMMIT(1 - buf);
        __syncthreads();
    }
#undef STAGE
#undef COMMIT

    // ---- epilogue: D vgpr r -> (row = r + g*8, col = fm) within 16x16 tile
#pragma unroll
    for (int i = 0; i < 4; ++i) {
        const int rowbase = m0 + wm + i * 16 + g * 8;
#pragma unroll
        for (int j = 0; j < 4; ++j) {
            const int col = n0 + wn + j * 16 + fm;
            const float bv = bias[col];
#pragma unroll
            for (int r = 0; r < 8; ++r) {
                float x = acc[i][j][r] + bv;
                C[(size_t)(rowbase + r) * N + col] = gelu_erf(x);
            }
        }
    }
}

// cell[b,t,h] = sum_{l : ids[b,l]==t} seq[b,l,h]  (deterministic l-order scan)
__global__ __launch_bounds__(256)
void segsum_kernel(const float* __restrict__ seq, const int* __restrict__ ids,
                   float* __restrict__ cell, int L, int H, int T) {
    __shared__ int sid[2048];
    const int t = blockIdx.x;
    const int b = blockIdx.y;
    const int* idp = ids + (size_t)b * L;
    for (int i = threadIdx.x; i < L; i += 256) sid[i] = idp[i];
    __syncthreads();

    const int h = threadIdx.x * 4;
    const float* sp = seq + (size_t)b * L * H + h;
    v4f sum = {0.0f, 0.0f, 0.0f, 0.0f};
    for (int l = 0; l < L; ++l) {
        if (sid[l] == t) {
            v4f v = *(const v4f*)(sp + (size_t)l * H);
            sum += v;
        }
    }
    *(v4f*)(cell + ((size_t)b * T + t) * H + h) = sum;
}

// out[b,l,h] = cellout[b, ids[b,l], h]
__global__ __launch_bounds__(256)
void gather_kernel(const float* __restrict__ cellout, const int* __restrict__ ids,
                   float* __restrict__ out, int L, int H, int T) {
    const int row = blockIdx.x;          // b*L + l
    const int b = row / L;
    const int l = row - b * L;
    const int t = ids[(size_t)b * L + l];
    const float* src = cellout + ((size_t)b * T + t) * H + threadIdx.x * 4;
    float* dst = out + (size_t)row * H + threadIdx.x * 4;
    *(v4f*)dst = *(const v4f*)src;
}

extern "C" void kernel_launch(void* const* d_in, const int* in_sizes, int n_in,
                              void* d_out, int out_size, void* d_ws, size_t ws_size,
                              hipStream_t stream) {
    const float* former = (const float*)d_in[0];   // [B,L,H] f32
    const float* hidden = (const float*)d_in[1];   // [B,L,H] f32
    const int*   ids    = (const int*)  d_in[2];   // [B,L] i32
    /* d_in[3] = attention_mask: unused by the reference */
    const float* W_down = (const float*)d_in[4];   // [2H,H] f32
    const float* b_down = (const float*)d_in[5];   // [H] f32
    const float* W_row  = (const float*)d_in[6];   // [H,H] f32
    const float* b_row  = (const float*)d_in[7];   // [H] f32
    float* out = (float*)d_out;

    const int H  = in_sizes[5];          // 1024
    const int BL = in_sizes[2];          // B*L = 16384
    const int L  = 2048;
    const int B  = BL / L;               // 8
    const int T  = 256;                  // max_length (reference constant)
    const int K1 = 2 * H;                // 2048
    const int M1 = BL;                   // 16384
    const int M2 = B * T;                // 2048
    const int K2 = H;                    // 1024

    // workspace layout (bytes): seq_out | cell_sum | cell_out | WdT | WrT
    char* ws = (char*)d_ws;
    float*  seq_out  = (float*)ws;
    float*  cell_sum = (float*)(ws + (size_t)M1 * H * 4);
    float*  cell_out = (float*)(ws + (size_t)M1 * H * 4 + (size_t)M2 * H * 4);
    bf16_t* WdT      = (bf16_t*)(ws + (size_t)M1 * H * 4 + 2 * (size_t)M2 * H * 4);
    bf16_t* WrT      = (bf16_t*)((char*)WdT + (size_t)K1 * H * 2);

    // 1) weight convert + transpose to [N][K] bf16
    convert_transpose_kernel<<<(H * K1) / 256, 256, 0, stream>>>(W_down, WdT, K1, H);
    convert_transpose_kernel<<<(H * K2) / 256, 256, 0, stream>>>(W_row,  WrT, K2, H);

    // 2) GEMM1: concat(former,hidden) @ W_down + b_down, GELU
    gemm_bias_gelu_kernel<<<dim3(M1 / TM, H / TN), 256, 0, stream>>>(
        former, hidden, H, WdT, K1, b_down, seq_out, H);

    // 3) deterministic segment-sum into the table cells
    segsum_kernel<<<dim3(T, B), 256, 0, stream>>>(seq_out, ids, cell_sum, L, H, T);

    // 4) GEMM2: cell_sum @ W_row + b_row, GELU
    gemm_bias_gelu_kernel<<<dim3(M2 / TM, H / TN), 256, 0, stream>>>(
        cell_sum, cell_sum, K2, WrT, K2, b_row, cell_out, H);

    // 5) gather cells back to token positions
    gather_kernel<<<B * L, 256, 0, stream>>>(cell_out, ids, out, L, H, T);
}
